// GS_Sentence_48928267436467
// MI455X (gfx1250) — compile-verified, matrix operand order checked by reading, not run on hardware
//
#include <hip/hip_runtime.h>
#include <hip/hip_bf16.h>
#include <math.h>

// ---------------------------------------------------------------------------
// Problem dims
//   B=128, L=128, V=300 (padded to 320), H=256, G=4H=1024, NE=5 experts,
//   10 chains = (expert, direction).  M = L*B = 16384 rows.
// ---------------------------------------------------------------------------

typedef _Float16 h16;
typedef __attribute__((ext_vector_type(16))) _Float16 v16h;
typedef __attribute__((ext_vector_type(8)))  _Float16 v8h;
typedef __attribute__((ext_vector_type(8)))  float    v8f;

#define WMMA_F16(A_, B_, C_) \
  __builtin_amdgcn_wmma_f32_16x16x32_f16(false, (A_), false, (B_), (short)0, (C_), false, false)

__device__ __forceinline__ v8f vzero8f() {
  v8f v;
#pragma unroll
  for (int i = 0; i < 8; ++i) v[i] = 0.f;
  return v;
}

// Load a 16x32 f16 fragment (A or B operand, both stored row-major [row][K])
// per CDNA5 ISA 7.12.2: lane l (0..15) holds row l, K = {0..7, 16..23};
// lane l+16 holds row l, K = {8..15, 24..31}.  Two 16-byte loads per lane.
__device__ __forceinline__ v16h frag_ld(const h16* __restrict__ base, int ld) {
  const int lane = threadIdx.x & 31;
  const int row  = lane & 15;
  const int ko   = (lane >> 4) << 3; // 0 or 8
  const h16* p = base + (size_t)row * ld + ko;
  v8h lo = *(const v8h*)(p);
  v8h hi = *(const v8h*)(p + 16);
  v16h r;
#pragma unroll
  for (int i = 0; i < 8; ++i) { r[i] = lo[i]; r[i + 8] = hi[i]; }
  return r;
}

__device__ __forceinline__ float sigmoidf_(float x) { return 1.f / (1.f + __expf(-x)); }

// ---------------------------------------------------------------------------
// Workspace layout (bytes).  Total = 616,366,080 (~588 MiB)
// ---------------------------------------------------------------------------
static const size_t OFF_XP    = 0;                          // f16 [16384][320]   x packed, row = l*128+b
static const size_t OFF_WIH   = OFF_XP    + 10485760;       // f16 [10][1024][320]
static const size_t OFF_WHH   = OFF_WIH   + 6553600;        // f16 [10][1024][256]
static const size_t OFF_ATTW  = OFF_WHH   + 5242880;        // f16 [5][512][512]
static const size_t OFF_XG    = OFF_ATTW  + 2621440;        // f16 [10][128][8][64][32][8]  WMMA-fragment-major
static const size_t OFF_HSEQF = OFF_XG    + 335544320;      // f32 [5][16384][512]    row = b*128+l
static const size_t OFF_HSEQH = OFF_HSEQF + 167772160;      // f16 [5][16384][512]
static const size_t OFF_SPART = OFF_HSEQH + 83886080;       // f32 [5][16384][8]  per-wave score partials
static const size_t OFF_AB    = OFF_SPART + 2621440;        // f32 [5][16384]     softmax weights
static const size_t OFF_OUTW  = OFF_AB    + 327680;         // f32 [5][128][512]  attention outputs

// ---------------------------------------------------------------------------
// Packing / conversion kernels
// ---------------------------------------------------------------------------
__global__ void k_pack_x(const float* __restrict__ x, h16* __restrict__ xp) {
  const int n = 16384 * 320;
  for (int i = blockIdx.x * blockDim.x + threadIdx.x; i < n; i += gridDim.x * blockDim.x) {
    int vp = i % 320;
    int m  = i / 320;          // m = l*128 + b
    int l  = m >> 7;
    int b  = m & 127;
    float v = (vp < 300) ? x[((size_t)((b << 7) + l)) * 300 + vp] : 0.f;  // x[b][l][v]
    xp[i] = (h16)v;
  }
}

__global__ void k_pack_wih(const float* __restrict__ wf, const float* __restrict__ wb,
                           h16* __restrict__ o) {
  const int n = 10 * 1024 * 320;
  for (int i = blockIdx.x * blockDim.x + threadIdx.x; i < n; i += gridDim.x * blockDim.x) {
    int vp = i % 320;
    int g  = (i / 320) & 1023;
    int c  = i / (320 * 1024);
    int e = c >> 1, dir = c & 1;
    const float* src = dir ? wb : wf;
    float v = (vp < 300) ? src[((size_t)e * 1024 + g) * 300 + vp] : 0.f;
    o[i] = (h16)v;
  }
}

__global__ void k_pack_whh(const float* __restrict__ wf, const float* __restrict__ wb,
                           h16* __restrict__ o) {
  const int n = 10 * 1024 * 256;
  for (int i = blockIdx.x * blockDim.x + threadIdx.x; i < n; i += gridDim.x * blockDim.x) {
    int k = i & 255;
    int g = (i >> 8) & 1023;
    int c = i >> 18;
    int e = c >> 1, dir = c & 1;
    const float* src = dir ? wb : wf;
    o[i] = (h16)src[((size_t)e * 1024 + g) * 256 + k];
  }
}

__global__ void k_pack_attw(const float* __restrict__ a, h16* __restrict__ o) {
  const int n = 5 * 512 * 512;
  for (int i = blockIdx.x * blockDim.x + threadIdx.x; i < n; i += gridDim.x * blockDim.x)
    o[i] = (h16)a[i];
}

// ---------------------------------------------------------------------------
// xg = x @ wih^T + bias   for 10 chains.  M=16384 (t,b), N=1024, K=320.
// One wave per 16x64 output strip.  Backward chains read time-reversed rows.
// Output is stored WMMA-fragment-major:
//   xg[c][t][mtile(8)][ntile(64)][lane(32)][r(8)]   (16B per lane per tile)
// so the LSTM kernel can init accumulators with one v8h load per gate tile.
// ---------------------------------------------------------------------------
__global__ __launch_bounds__(32) void k_xg(const h16* __restrict__ xp,
                                           const h16* __restrict__ wih16,
                                           const float* __restrict__ b_f,
                                           const float* __restrict__ b_b,
                                           h16* __restrict__ xg) {
  const int mt = blockIdx.x;           // 0..1023  (t = mt>>3, mtile = mt&7)
  const int nb = blockIdx.y;           // 0..15  (64 cols each)
  const int c  = blockIdx.z;           // chain
  const int e = c >> 1, dir = c & 1;
  const int lane   = threadIdx.x & 31;
  const int colsel = lane & 15;
  const int m0 = mt * 16;
  const int t   = mt >> 3;             // all 16 rows of a tile share t
  const int mfr = mt & 7;
  const int b0  = m0 & 127;
  const int lsrc = dir ? (127 - t) : t;
  const h16* A0 = xp + ((size_t)lsrc * 128 + b0) * 320;
  const h16* W0 = wih16 + (size_t)c * 1024 * 320 + (size_t)(nb * 64) * 320;
  const float* bias = (dir ? b_b : b_f) + (size_t)e * 1024 + nb * 64;

  v8f acc[4];
#pragma unroll
  for (int j = 0; j < 4; ++j) acc[j] = vzero8f();

  for (int kt = 0; kt < 10; ++kt) {
    v16h a = frag_ld(A0 + kt * 32, 320);
#pragma unroll
    for (int j = 0; j < 4; ++j) {
      v16h bfr = frag_ld(W0 + (size_t)(j * 16) * 320 + kt * 32, 320);
      acc[j] = WMMA_F16(a, bfr, acc[j]);
    }
  }

  // fragment-major store: tile (t, mfr, tn = nb*4+j), 8 contiguous f16 / lane
  h16* out0 = xg + ((((size_t)c * 128 + t) * 8 + mfr) * 64 + (size_t)nb * 4) * 256
                 + (size_t)lane * 8;
#pragma unroll
  for (int j = 0; j < 4; ++j) {
    float bv = bias[j * 16 + colsel];
    v8h st;
#pragma unroll
    for (int r = 0; r < 8; ++r) st[r] = (h16)(acc[j][r] + bv);
    *(v8h*)(out0 + (size_t)j * 256) = st;
  }
}

// ---------------------------------------------------------------------------
// LSTM recurrence: one workgroup per chain, h-state (128x256 f16) in LDS.
// 32 waves; wave owns 4 (m,n) tile-pairs, each pair needs gate tiles at
// n, n+256, n+512, n+768 so i/f/g/o land in the same lanes as the cell state.
// Accumulator init = one 16B vector load per gate tile (fragment-major xg).
// ---------------------------------------------------------------------------
__global__ __launch_bounds__(1024) void k_lstm(const h16* __restrict__ xg,
                                               const h16* __restrict__ whh16,
                                               float* __restrict__ hseqf,
                                               h16* __restrict__ hseqh) {
  extern __shared__ h16 smem[];        // 128 x 256 current h (64 KB)
  const int c = blockIdx.x;
  const int e = c >> 1, dir = c & 1;
  const int tid  = threadIdx.x;
  const int lane = tid & 31;
  const int w    = tid >> 5;
  const int colsel = lane & 15;
  const int rhalf  = (lane >> 4) * 8;

  const h16* whhc = whh16 + (size_t)c * 1024 * 256;
  const h16* xgc0 = xg + (size_t)c * 128 * 8 * 64 * 256;   // per-chain fragment base
  float* hf_e = hseqf + (size_t)e * 16384 * 512;
  h16*   hh_e = hseqh + (size_t)e * 16384 * 512;

  for (int i = tid; i < 128 * 256; i += 1024) smem[i] = (h16)0.f;

  float creg[4][8];
#pragma unroll
  for (int p = 0; p < 4; ++p)
#pragma unroll
    for (int r = 0; r < 8; ++r) creg[p][r] = 0.f;

  __syncthreads();

  for (int t = 0; t < 128; ++t) {
    const h16* xgt = xgc0 + (size_t)t * (8 * 64 * 256);
    const int lpos = dir ? (127 - t) : t;
    float hval[4][8];

#pragma unroll 1
    for (int tp = 0; tp < 4; ++tp) {
      const int p = w * 4 + tp;
      const int m = p >> 4, n = p & 15;
      const int b0 = m * 16;
      const int nI = n * 16, nF = 256 + n * 16, nG = 512 + n * 16, nO = 768 + n * 16;

      // init accumulators from fragment-major xg (gate g -> tile g*16+n)
      const h16* fb = xgt + ((size_t)m * 64) * 256 + (size_t)lane * 8;
      v8h rI = *(const v8h*)(fb + (size_t)(n)      * 256);
      v8h rF = *(const v8h*)(fb + (size_t)(16 + n) * 256);
      v8h rG = *(const v8h*)(fb + (size_t)(32 + n) * 256);
      v8h rO = *(const v8h*)(fb + (size_t)(48 + n) * 256);
      v8f aI, aF, aG, aO;
#pragma unroll
      for (int r = 0; r < 8; ++r) {
        aI[r] = (float)rI[r];
        aF[r] = (float)rF[r];
        aG[r] = (float)rG[r];
        aO[r] = (float)rO[r];
      }
      // g += h @ whh^T   (K = 256, 8 WMMA steps, 4 gate columns per A-frag)
      for (int kt = 0; kt < 8; ++kt) {
        v16h af = frag_ld(&smem[b0 * 256 + kt * 32], 256);
        v16h bI = frag_ld(whhc + (size_t)nI * 256 + kt * 32, 256);
        v16h bF = frag_ld(whhc + (size_t)nF * 256 + kt * 32, 256);
        v16h bG = frag_ld(whhc + (size_t)nG * 256 + kt * 32, 256);
        v16h bO = frag_ld(whhc + (size_t)nO * 256 + kt * 32, 256);
        aI = WMMA_F16(af, bI, aI);
        aF = WMMA_F16(af, bF, aF);
        aG = WMMA_F16(af, bG, aG);
        aO = WMMA_F16(af, bO, aO);
      }
      // cell / hidden update (register-local, same lanes as accumulators)
#pragma unroll
      for (int r = 0; r < 8; ++r) {
        float cv = sigmoidf_(aF[r]) * creg[tp][r] + sigmoidf_(aI[r]) * tanhf(aG[r]);
        creg[tp][r] = cv;
        hval[tp][r] = sigmoidf_(aO[r]) * tanhf(cv);
      }
    }

    __syncthreads();   // all reads of h(t-1) complete

#pragma unroll 1
    for (int tp = 0; tp < 4; ++tp) {
      const int p = w * 4 + tp;
      const int m = p >> 4, n = p & 15;
      const int b0 = m * 16;
      const int j  = n * 16 + colsel;
#pragma unroll
      for (int r = 0; r < 8; ++r) {
        const int b = b0 + rhalf + r;
        float hv = hval[tp][r];
        smem[b * 256 + j] = (h16)hv;
        size_t orow = ((size_t)b * 128 + lpos) * 512 + dir * 256 + j;
        hf_e[orow] = hv;
        hh_e[orow] = (h16)hv;
      }
    }

    __syncthreads();   // h(t) visible for next step
  }
}

// ---------------------------------------------------------------------------
// Fused attention score: s = tanh(h @ attW^T + attb) . attv
// M=16384, N=K=512.  8 waves cover N; deterministic per-wave partials.
// ---------------------------------------------------------------------------
__global__ __launch_bounds__(256) void k_att_s(const h16* __restrict__ hseqh,
                                               const h16* __restrict__ attw16,
                                               const float* __restrict__ attb,
                                               const float* __restrict__ attv,
                                               float* __restrict__ spart) {
  const int mt = blockIdx.x;           // 0..1023
  const int e  = blockIdx.y;           // 0..4
  const int tid  = threadIdx.x;
  const int lane = tid & 31;
  const int w    = tid >> 5;           // 0..7, 64 cols each
  const int colsel = lane & 15;
  const int rhalf  = (lane >> 4) * 8;
  const int m0 = mt * 16;

  const h16* A0 = hseqh + ((size_t)e * 16384 + m0) * 512;
  const h16* W0 = attw16 + (size_t)e * 512 * 512 + (size_t)(w * 64) * 512;

  v8f acc[4];
#pragma unroll
  for (int j = 0; j < 4; ++j) acc[j] = vzero8f();

  for (int kt = 0; kt < 16; ++kt) {
    v16h a = frag_ld(A0 + kt * 32, 512);
#pragma unroll
    for (int j = 0; j < 4; ++j) {
      v16h bfr = frag_ld(W0 + (size_t)(j * 16) * 512 + kt * 32, 512);
      acc[j] = WMMA_F16(a, bfr, acc[j]);
    }
  }

  float part[8];
#pragma unroll
  for (int r = 0; r < 8; ++r) part[r] = 0.f;
#pragma unroll
  for (int j = 0; j < 4; ++j) {
    const int n = w * 64 + j * 16 + colsel;
    const float bb = attb[(size_t)e * 512 + n];
    const float av = attv[(size_t)e * 512 + n];
#pragma unroll
    for (int r = 0; r < 8; ++r) part[r] += tanhf(acc[j][r] + bb) * av;
  }
#pragma unroll
  for (int r = 0; r < 8; ++r) {
    for (int msk = 8; msk >= 1; msk >>= 1) part[r] += __shfl_xor(part[r], msk);
    if (colsel == 0)
      spart[((size_t)e * 16384 + m0 + rhalf + r) * 8 + w] = part[r];
  }
}

// ---------------------------------------------------------------------------
// Softmax over L per (e,b): reduce the 8 per-wave partials, then softmax.
// ---------------------------------------------------------------------------
__global__ __launch_bounds__(128) void k_softmax(const float* __restrict__ spart,
                                                 float* __restrict__ abuf) {
  __shared__ float red[128];
  const int eb = blockIdx.x;           // e*128 + b
  const int l  = threadIdx.x;
  const float* sp = spart + ((size_t)eb * 128 + l) * 8;
  float v = 0.f;
#pragma unroll
  for (int k = 0; k < 8; ++k) v += sp[k];

  red[l] = v; __syncthreads();
  for (int s = 64; s > 0; s >>= 1) { if (l < s) red[l] = fmaxf(red[l], red[l + s]); __syncthreads(); }
  float mx = red[0]; __syncthreads();
  float ev = __expf(v - mx);
  red[l] = ev; __syncthreads();
  for (int s = 64; s > 0; s >>= 1) { if (l < s) red[l] += red[l + s]; __syncthreads(); }
  abuf[(size_t)eb * 128 + l] = ev / red[0];
}

// ---------------------------------------------------------------------------
// out[e][b][j] = sum_l a[e][b][l] * h[e][b][l][j]
// ---------------------------------------------------------------------------
__global__ __launch_bounds__(128) void k_attout(const float* __restrict__ abuf,
                                                const float* __restrict__ hseqf,
                                                float* __restrict__ outw) {
  __shared__ float as_[128];
  const int b = blockIdx.x, e = blockIdx.y, tid = threadIdx.x;
  as_[tid] = abuf[((size_t)e * 128 + b) * 128 + tid];
  __syncthreads();
  const float* h0 = hseqf + ((size_t)e * 16384 + (size_t)b * 128) * 512;
#pragma unroll 1
  for (int jj = 0; jj < 4; ++jj) {
    const int j = tid + jj * 128;
    float acc = 0.f;
    for (int l = 0; l < 128; ++l) acc += as_[l] * h0[(size_t)l * 512 + j];
    outw[((size_t)e * 128 + b) * 512 + j] = acc;
  }
}

// ---------------------------------------------------------------------------
// Final: select expert by z, concat [sel, general], FC to 2, emit outputs.
// d_out layout: final (128*2) | general (128*512) | specifics (4*128*512)
// ---------------------------------------------------------------------------
__global__ __launch_bounds__(128) void k_final(const float* __restrict__ outw,
                                               const int* __restrict__ z,
                                               const float* __restrict__ fcW,
                                               const float* __restrict__ fcb,
                                               float* __restrict__ dout) {
  __shared__ float cat[1024];
  __shared__ float red[128];
  const int b = blockIdx.x, tid = threadIdx.x;
  int se = z[b] + 1;
  if (se < 1) se = 1;
  if (se > 4) se = 4;
  const float* osel = outw + ((size_t)se * 128 + b) * 512;
  const float* ogen = outw + (size_t)b * 512;
  for (int d = tid; d < 512; d += 128) { cat[d] = osel[d]; cat[512 + d] = ogen[d]; }
  __syncthreads();
  for (int o = 0; o < 2; ++o) {
    float p = 0.f;
    for (int d = tid; d < 1024; d += 128) p += fcW[(size_t)o * 1024 + d] * cat[d];
    red[tid] = p; __syncthreads();
    for (int s = 64; s > 0; s >>= 1) { if (tid < s) red[tid] += red[tid + s]; __syncthreads(); }
    if (tid == 0) dout[b * 2 + o] = red[0] + fcb[o];
    __syncthreads();
  }
  for (int j = tid; j < 512; j += 128)
    dout[256 + (size_t)b * 512 + j] = ogen[j];
  for (int k = 0; k < 4; ++k)
    for (int j = tid; j < 512; j += 128)
      dout[256 + 65536 + ((size_t)k * 128 + b) * 512 + j] =
          outw[((size_t)(k + 1) * 128 + b) * 512 + j];
}

// ---------------------------------------------------------------------------
extern "C" void kernel_launch(void* const* d_in, const int* in_sizes, int n_in,
                              void* d_out, int out_size, void* d_ws, size_t ws_size,
                              hipStream_t stream) {
  const float* x     = (const float*)d_in[0];
  const int*   z     = (const int*)  d_in[1];
  const float* wih_f = (const float*)d_in[2];
  const float* whh_f = (const float*)d_in[3];
  const float* b_f   = (const float*)d_in[4];
  const float* wih_b = (const float*)d_in[5];
  const float* whh_b = (const float*)d_in[6];
  const float* b_b   = (const float*)d_in[7];
  const float* attW  = (const float*)d_in[8];
  const float* attb  = (const float*)d_in[9];
  const float* attv  = (const float*)d_in[10];
  const float* fcW   = (const float*)d_in[11];
  const float* fcb   = (const float*)d_in[12];
  (void)in_sizes; (void)n_in; (void)out_size; (void)ws_size;

  char* ws = (char*)d_ws;
  h16*   xp     = (h16*)(ws + OFF_XP);
  h16*   wih16  = (h16*)(ws + OFF_WIH);
  h16*   whh16  = (h16*)(ws + OFF_WHH);
  h16*   attw16 = (h16*)(ws + OFF_ATTW);
  h16*   xg16   = (h16*)(ws + OFF_XG);
  float* hseqf  = (float*)(ws + OFF_HSEQF);
  h16*   hseqh  = (h16*)(ws + OFF_HSEQH);
  float* spart  = (float*)(ws + OFF_SPART);
  float* abuf   = (float*)(ws + OFF_AB);
  float* outw   = (float*)(ws + OFF_OUTW);

  k_pack_x  <<<(16384 * 320 + 255) / 256, 256, 0, stream>>>(x, xp);
  k_pack_wih<<<(10 * 1024 * 320 + 255) / 256, 256, 0, stream>>>(wih_f, wih_b, wih16);
  k_pack_whh<<<(10 * 1024 * 256 + 255) / 256, 256, 0, stream>>>(whh_f, whh_b, whh16);
  k_pack_attw<<<(5 * 512 * 512 + 255) / 256, 256, 0, stream>>>(attW, attw16);

  k_xg<<<dim3(1024, 16, 10), 32, 0, stream>>>(xp, wih16, b_f, b_b, xg16);

  k_lstm<<<10, 1024, 128 * 256 * (int)sizeof(h16), stream>>>(xg16, whh16, hseqf, hseqh);

  k_att_s<<<dim3(1024, 5), 256, 0, stream>>>(hseqh, attw16, attb, attv, spart);
  k_softmax<<<640, 128, 0, stream>>>(spart, abuf);
  k_attout<<<dim3(128, 5), 128, 0, stream>>>(abuf, hseqf, outw);
  k_final<<<128, 128, 0, stream>>>(outw, z, fcW, fcb, (float*)d_out);
}
